// Attention_10084583211250
// MI455X (gfx1250) — matrix operand order repaired and checked by reading, not compiled, not run on hardware
//
#include <hip/hip_runtime.h>

typedef __bf16 bf16_t;
typedef __attribute__((ext_vector_type(16))) __bf16 v16bf;
typedef __attribute__((ext_vector_type(8)))  __bf16 v8bf;
typedef __attribute__((ext_vector_type(8)))  float  v8f;

constexpr int Bb  = 2;
constexpr int Ss  = 2048;
constexpr int Dd  = 2048;
constexpr int Hh  = 32;
constexpr int KVH = 8;
constexpr int HD  = 64;
constexpr int GRP = Hh / KVH;           // 4
constexpr float SCALE = 0.125f;          // 64^-0.5

__device__ __forceinline__ v8f wmma_bf16(v16bf a, v16bf b, v8f c) {
    return __builtin_amdgcn_wmma_f32_16x16x32_bf16(
        false, a, false, b, (short)0, c, false, false);
}

// Build a v16bf A-operand from two contiguous 16-byte chunks.
__device__ __forceinline__ v16bf load2(const bf16_t* p0, const bf16_t* p1) {
    v8bf lo = *(const v8bf*)p0;
    v8bf hi = *(const v8bf*)p1;
    v16bf r;
#pragma unroll
    for (int i = 0; i < 8; ++i) { r[i] = lo[i]; r[i + 8] = hi[i]; }
    return r;
}

// ---------------- elementwise prep kernels ----------------

__global__ void cvt_bf16_kernel(const float* __restrict__ x, bf16_t* __restrict__ y, int n) {
    int i = blockIdx.x * blockDim.x + threadIdx.x;
    if (i < n) y[i] = (bf16_t)x[i];
}

// W[k,n] (row-major KxN) -> Wt[n,k] bf16
__global__ void transpose_w_kernel(const float* __restrict__ w, bf16_t* __restrict__ wt,
                                   int K, int N) {
    int i = blockIdx.x * blockDim.x + threadIdx.x;
    if (i >= K * N) return;
    int n = i / K, k = i % K;
    wt[(size_t)n * K + k] = (bf16_t)w[(size_t)k * N + n];
}

// x: fp32 [B*S, nh*HD]  ->  y: bf16 [B, nh, S, HD] with RoPE applied
__global__ void rope_kernel(const float* __restrict__ x, const float* __restrict__ cosT,
                            const float* __restrict__ sinT, bf16_t* __restrict__ y, int nh) {
    int i = blockIdx.x * blockDim.x + threadIdx.x;
    int total = Bb * Ss * nh * HD;
    if (i >= total) return;
    int d = i % HD;
    int h = (i / HD) % nh;
    int s = (i / (HD * nh)) % Ss;
    int b = i / (HD * nh * Ss);
    size_t base = ((size_t)(b * Ss + s)) * (nh * HD) + h * HD;
    float v = x[base + d];
    float rot = (d < HD / 2) ? -x[base + d + HD / 2] : x[base + d - HD / 2];
    float out = v * cosT[s * HD + d] + rot * sinT[s * HD + d];
    y[(((size_t)b * nh + h) * Ss + s) * HD + d] = (bf16_t)out;
}

// v: fp32 [B*S, KVH*HD] -> vt: bf16 [B, KVH, HD, S]  (transposed for AV wmma A-operand)
__global__ void vtrans_kernel(const float* __restrict__ v, bf16_t* __restrict__ vt) {
    int i = blockIdx.x * blockDim.x + threadIdx.x;
    int total = Bb * Ss * KVH * HD;
    if (i >= total) return;
    int d  = i % HD;
    int kv = (i / HD) % KVH;
    int s  = (i / (HD * KVH)) % Ss;
    int b  = i / (HD * KVH * Ss);
    vt[(((size_t)b * KVH + kv) * HD + d) * Ss + s] =
        (bf16_t)v[((size_t)(b * Ss + s)) * (KVH * HD) + kv * HD + d];
}

// ---------------- WMMA GEMM: C[M,N] fp32 = A[M,K] bf16 x Bt[N,K] bf16 ----------------
// One wave computes a 32x64 tile (2 m-subtiles share each B load -> B traffic halved).
// Waves within a block share the same n-range with consecutive m-tiles, so the
// dominant B (weight) stream hits the WGP cache across the block's 4 waves.

__global__ __launch_bounds__(128) void gemm_bf16_kernel(const bf16_t* __restrict__ A,
                                                        const bf16_t* __restrict__ Bt,
                                                        float* __restrict__ C,
                                                        int M, int N, int K) {
    const int lane = threadIdx.x & 31;
    const int wid  = blockIdx.x * (blockDim.x >> 5) + (threadIdx.x >> 5);
    const int tilesM = M >> 5;               // 32-row tiles
    const int mt = wid % tilesM;
    const int nt = wid / tilesM;
    if (nt * 64 >= N) return;
    const int col  = lane & 15;
    const int half = lane >> 4;

    const bf16_t* Arow0 = A + (size_t)(mt * 32 + col) * K;
    const bf16_t* Arow1 = A + (size_t)(mt * 32 + 16 + col) * K;
    const bf16_t* B0    = Bt + (size_t)(nt * 64 + col) * K;

    v8f acc0[4] = {};
    v8f acc1[4] = {};
    for (int kb = 0; kb < K; kb += 32) {
        v16bf a0 = load2(Arow0 + kb + half * 8, Arow0 + kb + 16 + half * 8);
        v16bf a1 = load2(Arow1 + kb + half * 8, Arow1 + kb + 16 + half * 8);
#pragma unroll
        for (int j = 0; j < 4; ++j) {
            v16bf b = load2(B0 + (size_t)j * 16 * K + kb + half * 8, B0 + (size_t)j * 16 * K + kb + 16 + half * 8);
            acc0[j] = wmma_bf16(a0, b, acc0[j]);
            acc1[j] = wmma_bf16(a1, b, acc1[j]);
        }
    }
    // C layout per 16x16 tile: M = r + 8*half, N = col
#pragma unroll
    for (int j = 0; j < 4; ++j)
#pragma unroll
        for (int r = 0; r < 8; ++r) {
            C[(size_t)(mt * 32 + r + 8 * half) * N + nt * 64 + j * 16 + col]      = acc0[j][r];
            C[(size_t)(mt * 32 + 16 + r + 8 * half) * N + nt * 64 + j * 16 + col] = acc1[j][r];
        }
}

// ---------------- Flash attention (Sᵀ = K·Qᵀ trick, Oᵀ = Vᵀ·Pᵀ) ----------------
// One wave per 16-query tile; 32 keys per iteration; 8 WMMAs per iteration.
// Per-lane online softmax (lane == query column), no LDS transpose needed.

__global__ __launch_bounds__(128) void attn_kernel(const bf16_t* __restrict__ Qb,
                                                   const bf16_t* __restrict__ Kb,
                                                   const bf16_t* __restrict__ Vt,
                                                   bf16_t* __restrict__ Ob) {
    const int lane = threadIdx.x & 31;
    const int wid  = blockIdx.x * (blockDim.x >> 5) + (threadIdx.x >> 5);
    const int qtiles = Ss / 16;
    const int qt = wid % qtiles;
    const int h  = (wid / qtiles) % Hh;
    const int b  = wid / (qtiles * Hh);
    const int kv = h / GRP;
    const int qbase = qt * 16;
    const int col  = lane & 15;
    const int half = lane >> 4;
    const int qglob = qbase + col;

    // Q B-operand (reused across all key tiles): lane owns query column `col`
    const bf16_t* Qrow = Qb + (((size_t)(b * Hh + h) * Ss) + qbase + col) * HD;
    v16bf bq0 = load2(Qrow + half * 8, Qrow + 16 + half * 8);        // dims 0..31
    v16bf bq1 = load2(Qrow + 32 + half * 8, Qrow + 48 + half * 8);   // dims 32..63

    const bf16_t* Kbase = Kb + ((size_t)(b * KVH + kv) * Ss) * HD;
    const bf16_t* Vbase = Vt + ((size_t)(b * KVH + kv) * HD) * Ss;

    float m = -1e30f, l = 0.f;
    v8f acc0 = {}, acc1 = {}, acc2 = {}, acc3 = {};

    for (int kb = 0; kb < qbase + 16; kb += 32) {
        // ---- scores: Sᵀ[key, query] ----
        int kr0 = kb + col;      if (kr0 > Ss - 1) kr0 = Ss - 1;
        int kr1 = kb + 16 + col; if (kr1 > Ss - 1) kr1 = Ss - 1;
        const bf16_t* K0 = Kbase + (size_t)kr0 * HD;
        const bf16_t* K1 = Kbase + (size_t)kr1 * HD;

        v8f c0 = {}, c1 = {};
        c0 = wmma_bf16(load2(K0 + half * 8,      K0 + 16 + half * 8), bq0, c0);
        c0 = wmma_bf16(load2(K0 + 32 + half * 8, K0 + 48 + half * 8), bq1, c0);
        c1 = wmma_bf16(load2(K1 + half * 8,      K1 + 16 + half * 8), bq0, c1);
        c1 = wmma_bf16(load2(K1 + 32 + half * 8, K1 + 48 + half * 8), bq1, c1);

        // ---- causal mask + scale; per-lane (= per-query) online softmax ----
        float p0[8], p1[8];
        float tmax = -1e30f;
#pragma unroll
        for (int r = 0; r < 8; ++r) {
            int k0 = kb + r + 8 * half;
            int k1 = kb + 16 + r + 8 * half;
            float s0 = (k0 <= qglob) ? c0[r] * SCALE : -1e30f;
            float s1 = (k1 <= qglob) ? c1[r] * SCALE : -1e30f;
            p0[r] = s0; p1[r] = s1;
            tmax = fmaxf(tmax, fmaxf(s0, s1));
        }
        tmax = fmaxf(tmax, __shfl_xor(tmax, 16));
        float mnew = fmaxf(m, tmax);
        float corr = __expf(m - mnew);
        float lsum = 0.f;
#pragma unroll
        for (int r = 0; r < 8; ++r) {
            p0[r] = (p0[r] > -1e29f) ? __expf(p0[r] - mnew) : 0.f;
            p1[r] = (p1[r] > -1e29f) ? __expf(p1[r] - mnew) : 0.f;
            lsum += p0[r] + p1[r];
        }
        lsum += __shfl_xor(lsum, 16);
        l = l * corr + lsum;
        m = mnew;
        acc0 = acc0 * corr; acc1 = acc1 * corr; acc2 = acc2 * corr; acc3 = acc3 * corr;

        // ---- Pᵀ (C layout) is already in B-operand layout; lane keeps its own values (t0/t1 unused) ----
        v16bf bp;
#pragma unroll
        for (int e = 0; e < 8; ++e) {
            float t0 = __shfl_xor(p0[e], 16);
            float t1 = __shfl_xor(p1[e], 16);
            float lo = p0[e];   // elements 0..7  : keys half*8 + e
            float hi = p1[e];   // elements 8..15 : keys 16 + half*8 + e
            bp[e]     = (bf16_t)lo;
            bp[e + 8] = (bf16_t)hi;
        }

        // ---- Oᵀ += Vᵀ · Pᵀ  (4 dim-groups of 16) ----
        const bf16_t* V0 = Vbase + (size_t)(0 * 16 + col) * Ss + kb;
        const bf16_t* V1 = Vbase + (size_t)(1 * 16 + col) * Ss + kb;
        const bf16_t* V2 = Vbase + (size_t)(2 * 16 + col) * Ss + kb;
        const bf16_t* V3 = Vbase + (size_t)(3 * 16 + col) * Ss + kb;
        acc0 = wmma_bf16(load2(V0 + half * 8, V0 + 16 + half * 8), bp, acc0);
        acc1 = wmma_bf16(load2(V1 + half * 8, V1 + 16 + half * 8), bp, acc1);
        acc2 = wmma_bf16(load2(V2 + half * 8, V2 + 16 + half * 8), bp, acc2);
        acc3 = wmma_bf16(load2(V3 + half * 8, V3 + 16 + half * 8), bp, acc3);
    }

    float inv = 1.f / l;
    // Ob: bf16 [B*S, H*HD]; lane owns query row qglob, dims g*16 + r + 8*half
    bf16_t* Orow = Ob + ((size_t)(b * Ss) + qbase + col) * (Hh * HD) + h * HD;
    v8f accs[4] = {acc0, acc1, acc2, acc3};
#pragma unroll
    for (int g = 0; g < 4; ++g) {
#pragma unroll
        for (int r = 0; r < 8; r += 2) {
            union { bf16_t h2[2]; unsigned u; } pk;
            pk.h2[0] = (bf16_t)(accs[g][r] * inv);
            pk.h2[1] = (bf16_t)(accs[g][r + 1] * inv);
            *(unsigned*)(Orow + g * 16 + 8 * half + r) = pk.u;
        }
    }
}

// ---------------- host launcher ----------------

extern "C" void kernel_launch(void* const* d_in, const int* in_sizes, int n_in,
                              void* d_out, int out_size, void* d_ws, size_t ws_size,
                              hipStream_t stream) {
    (void)in_sizes; (void)n_in; (void)out_size; (void)ws_size;
    const float* hs   = (const float*)d_in[0];
    // d_in[1] = attention_mask (causal; applied analytically)
    const float* cosT = (const float*)d_in[2];
    const float* sinT = (const float*)d_in[3];
    const float* Wq   = (const float*)d_in[4];
    const float* Wk   = (const float*)d_in[5];
    const float* Wv   = (const float*)d_in[6];
    const float* Wo   = (const float*)d_in[7];
    float* out = (float*)d_out;

    char* ws = (char*)d_ws;
    size_t off = 0;
    auto alloc = [&](size_t bytes) -> void* {
        void* p = ws + off;
        off += (bytes + 255) & ~(size_t)255;
        return p;
    };

    const int MROWS = Bb * Ss;                 // 4096
    bf16_t* Xb  = (bf16_t*)alloc((size_t)MROWS * Dd * 2);
    bf16_t* WqT = (bf16_t*)alloc((size_t)Dd * (Hh * HD) * 2);
    bf16_t* WkT = (bf16_t*)alloc((size_t)Dd * (KVH * HD) * 2);
    bf16_t* WvT = (bf16_t*)alloc((size_t)Dd * (KVH * HD) * 2);
    bf16_t* WoT = (bf16_t*)alloc((size_t)(Hh * HD) * Dd * 2);
    float*  Qf  = (float*) alloc((size_t)MROWS * (Hh * HD) * 4);
    float*  Kf  = (float*) alloc((size_t)MROWS * (KVH * HD) * 4);
    float*  Vf  = (float*) alloc((size_t)MROWS * (KVH * HD) * 4);
    bf16_t* Qb  = (bf16_t*)alloc((size_t)Bb * Hh * Ss * HD * 2);
    bf16_t* Kb  = (bf16_t*)alloc((size_t)Bb * KVH * Ss * HD * 2);
    bf16_t* Vt  = (bf16_t*)alloc((size_t)Bb * KVH * HD * Ss * 2 + 256); // pad: masked tail reads
    bf16_t* Ob  = (bf16_t*)alloc((size_t)MROWS * (Hh * HD) * 2);

    const int T = 256;
    // 1) conversions / transposes
    {
        int n = MROWS * Dd;
        cvt_bf16_kernel<<<n / T, T, 0, stream>>>(hs, Xb, n);
    }
    transpose_w_kernel<<<(Dd * Hh * HD) / T, T, 0, stream>>>(Wq, WqT, Dd, Hh * HD);
    transpose_w_kernel<<<(Dd * KVH * HD) / T, T, 0, stream>>>(Wk, WkT, Dd, KVH * HD);
    transpose_w_kernel<<<(Dd * KVH * HD) / T, T, 0, stream>>>(Wv, WvT, Dd, KVH * HD);
    transpose_w_kernel<<<(Hh * HD * Dd) / T, T, 0, stream>>>(Wo, WoT, Hh * HD, Dd);

    // 2) QKV projections (WMMA GEMM, 32x64 tiles)
    {
        int blocks = (MROWS / 32) * ((Hh * HD) / 64) / 4;
        gemm_bf16_kernel<<<blocks, 128, 0, stream>>>(Xb, WqT, Qf, MROWS, Hh * HD, Dd);
    }
    {
        int blocks = (MROWS / 32) * ((KVH * HD) / 64) / 4;
        gemm_bf16_kernel<<<blocks, 128, 0, stream>>>(Xb, WkT, Kf, MROWS, KVH * HD, Dd);
        gemm_bf16_kernel<<<blocks, 128, 0, stream>>>(Xb, WvT, Vf, MROWS, KVH * HD, Dd);
    }

    // 3) RoPE + layout changes
    rope_kernel<<<(Bb * Ss * Hh * HD) / T, T, 0, stream>>>(Qf, cosT, sinT, Qb, Hh);
    rope_kernel<<<(Bb * Ss * KVH * HD) / T, T, 0, stream>>>(Kf, cosT, sinT, Kb, KVH);
    vtrans_kernel<<<(Bb * Ss * KVH * HD) / T, T, 0, stream>>>(Vf, Vt);

    // 4) flash attention
    {
        int waves = Bb * Hh * (Ss / 16);   // 8192
        attn_kernel<<<waves / 4, 128, 0, stream>>>(Qb, Kb, Vt, Ob);
    }

    // 5) output projection
    {
        int blocks = (MROWS / 32) * (Dd / 64) / 4;
        gemm_bf16_kernel<<<blocks, 128, 0, stream>>>(Ob, WoT, out, MROWS, Dd, Dd);
    }
}
